// AstraGNNWrapper_50989851738658
// MI455X (gfx1250) — matrix operand order repaired and checked by reading, not compiled, not run on hardware
//
#include <hip/hip_runtime.h>
#include <stdint.h>

// ---------------- problem constants (from reference) ----------------
#define kN   50000
#define kE   1600000
#define kH   128
#define kITERS 3

#define PITCH_M  136        // 128 + 8 halfword pad for the m1 transpose tile
#define BLK_WAVES 8         // 256-thread blocks; wave = one 16-row tile

typedef __attribute__((ext_vector_type(16))) __bf16          v16bf;
typedef __attribute__((ext_vector_type(8)))  float           v8f;
typedef __attribute__((ext_vector_type(8)))  unsigned short  us8;
typedef __attribute__((ext_vector_type(16))) unsigned short  us16;

__device__ __forceinline__ unsigned short f2bf(float f) {
    union { float f; uint32_t u; } v; v.f = f;
    uint32_t u = v.u;
    uint32_t r = u + 0x7fffu + ((u >> 16) & 1u);   // RNE
    return (unsigned short)(r >> 16);
}

__device__ __forceinline__ v8f vzero8() {
    v8f z = {0.f,0.f,0.f,0.f,0.f,0.f,0.f,0.f};
    return z;
}

__device__ __forceinline__ v16bf combine_a(us8 a0, us8 a1) {
    us16 au = __builtin_shufflevector(a0, a1,
        0,1,2,3,4,5,6,7,8,9,10,11,12,13,14,15);
    return __builtin_bit_cast(v16bf, au);
}

// B fragment from the LDS weight buffer: frag nb, per-lane 32B chunk.
__device__ __forceinline__ v16bf lds_b_frag(const unsigned short* wbuf,
                                            int nb, int lane) {
    const us16* p = (const us16*)(wbuf + nb * 512 + lane * 16);
    return __builtin_bit_cast(v16bf, *p);
}

// Cooperative copy of one 8KB k-step slab of packed weights into LDS.
__device__ __forceinline__ void stage_weights(const unsigned short* __restrict__ Wp,
                                              unsigned short* wbuf,
                                              int kb, int tid) {
    const uint4* g = (const uint4*)(Wp + (size_t)kb * 4096);
    uint4* s = (uint4*)wbuf;
    s[2 * tid]     = g[2 * tid];
    s[2 * tid + 1] = g[2 * tid + 1];
}

// ---------------- weight pre-pack: fp32 [K,128] -> bf16 B-fragments ----------
// Packed order: frag f = (kb*8 + nb), then lane, then 16 bf16 -> the 8 frags of
// one k-step are one contiguous 8KB slab (cooperative-staging friendly).
__global__ __launch_bounds__(256) void pack_b_kernel(const float* __restrict__ W,
                                                     unsigned short* __restrict__ out,
                                                     int KB) {
    int t = blockIdx.x * 256 + threadIdx.x;       // one thread = (frag, lane)
    int total = KB * 8 * 32;
    if (t >= total) return;
    int lane = t & 31;
    int f    = t >> 5;
    int nb   = f & 7;
    int kb   = f >> 3;
    int n     = nb * 16 + (lane & 15);
    int kbase = kb * 32 + (lane >> 4) * 8;
    unsigned int buf[8];
#pragma unroll
    for (int i = 0; i < 8; ++i) {
        unsigned short lo = f2bf(W[(size_t)(kbase + 2 * i + (i >= 4 ? 8 : 0)    ) * kH + n]);
        unsigned short hi = f2bf(W[(size_t)(kbase + 2 * i + (i >= 4 ? 8 : 0) + 1) * kH + n]);
        buf[i] = (unsigned int)lo | ((unsigned int)hi << 16);
    }
    uint4* o = (uint4*)(out + ((size_t)t << 4));
    o[0] = make_uint4(buf[0], buf[1], buf[2], buf[3]);
    o[1] = make_uint4(buf[4], buf[5], buf[6], buf[7]);
}

// ---------------- h init: fp32 master + bf16 shadow -------------------------
__global__ __launch_bounds__(256) void init_h_kernel(const float* __restrict__ x,
                                                     float* __restrict__ h,
                                                     unsigned short* __restrict__ hbf) {
    size_t i = (size_t)blockIdx.x * 256 + threadIdx.x;
    if (i >= (size_t)kN * kH) return;
    float v = x[i];
    h[i] = v;
    hbf[i] = f2bf(v);
}

__global__ __launch_bounds__(256) void zero_kernel(float* __restrict__ p, size_t n) {
    size_t i = (size_t)blockIdx.x * 256 + threadIdx.x;
    if (i < n) p[i] = 0.f;
}

// ---------------- edge kernel: msg MLP + masked scatter-add -----------------
// 256 threads = 8 waves; each wave owns 16 edges. Weights staged per k-step
// into double-buffered LDS and shared by all 8 waves (W1+W2 read once per
// 128 edges). A fragments gathered straight from L2 (concat split by kb).
__global__ __launch_bounds__(256) void edge_kernel(
    const unsigned short* __restrict__ hbf,    // [N,128] bf16
    const int*            __restrict__ eidx,   // [2,E]
    const unsigned char*  __restrict__ emask,  // [E]
    const unsigned short* __restrict__ W1p,    // packed 8 k-slabs
    const float*          __restrict__ b1,
    const unsigned short* __restrict__ W2p,    // packed 4 k-slabs
    const float*          __restrict__ b2,
    float*                __restrict__ agg)    // [N,128]
{
    __shared__ __align__(32) unsigned short wbuf[2][4096];        // 2 x 8KB
    __shared__ __align__(16) unsigned short smM1[BLK_WAVES][16 * PITCH_M];

    const int tid  = threadIdx.x;
    const int wave = tid >> 5;
    const int lane = tid & 31;
    const int ebase = (blockIdx.x * BLK_WAVES + wave) * 16;
    const int kh  = lane >> 4;
    const int nlo = lane & 15;
    const int m   = lane & 15;

    const int e0   = ebase + m;
    const int srcI = eidx[e0];
    const int dstI = eidx[kE + e0];

    // ---- layer 1: [16x256] @ W1 -> [16x128] ----
    v8f acc[8];
#pragma unroll
    for (int nb = 0; nb < 8; ++nb) acc[nb] = vzero8();
#pragma unroll
    for (int kb = 0; kb < 8; ++kb) {
        const int buf = kb & 1;
        stage_weights(W1p, &wbuf[buf][0], kb, tid);
        __syncthreads();
        // A fragment direct from global: kb 0-3 -> h[src], kb 4-7 -> h[dst]
        const int rowI = (kb < 4) ? srcI : dstI;
        const unsigned short* ab =
            hbf + (size_t)rowI * kH + (kb & 3) * 32 + kh * 8;
        v16bf a = combine_a(*(const us8*)ab, *(const us8*)(ab + 16));
        // batch-load all 8 B fragments, then issue the 8 WMMAs
        v16bf bf[8];
#pragma unroll
        for (int nb = 0; nb < 8; ++nb) bf[nb] = lds_b_frag(&wbuf[buf][0], nb, lane);
#pragma unroll
        for (int nb = 0; nb < 8; ++nb) {
            acc[nb] = __builtin_amdgcn_wmma_f32_16x16x32_bf16(
                false, a, false, bf[nb], (short)0, acc[nb], false, false);
        }
    }
    // ReLU + bf16 -> per-wave LDS transpose tile
#pragma unroll
    for (int nb = 0; nb < 8; ++nb) {
        float bias = b1[nb * 16 + nlo];
#pragma unroll
        for (int r = 0; r < 8; ++r) {
            float x = fmaxf(acc[nb][r] + bias, 0.f);
            int mr = r + kh * 8;
            smM1[wave][mr * PITCH_M + nb * 16 + nlo] = f2bf(x);
        }
    }
    __syncthreads();

    // ---- layer 2: [16x128] @ W2 -> [16x128] ----
    v8f acc2[8];
#pragma unroll
    for (int nb = 0; nb < 8; ++nb) acc2[nb] = vzero8();
#pragma unroll
    for (int kb = 0; kb < 4; ++kb) {
        const int buf = kb & 1;
        stage_weights(W2p, &wbuf[buf][0], kb, tid);
        __syncthreads();
        const unsigned short* ab = &smM1[wave][m * PITCH_M + kb * 32 + kh * 8];
        v16bf a = combine_a(*(const us8*)ab, *(const us8*)(ab + 16));
        v16bf bf[8];
#pragma unroll
        for (int nb = 0; nb < 8; ++nb) bf[nb] = lds_b_frag(&wbuf[buf][0], nb, lane);
#pragma unroll
        for (int nb = 0; nb < 8; ++nb) {
            acc2[nb] = __builtin_amdgcn_wmma_f32_16x16x32_bf16(
                false, a, false, bf[nb], (short)0, acc2[nb], false, false);
        }
    }

    // ---- epilogue: + b2, edge mask, atomic scatter-add into agg[dst] ----
    float bias2[8];
#pragma unroll
    for (int nb = 0; nb < 8; ++nb) bias2[nb] = b2[nb * 16 + nlo];
#pragma unroll
    for (int r = 0; r < 8; ++r) {
        int mr = r + kh * 8;
        int e  = ebase + mr;
        if (emask[e]) {
            int d = eidx[kE + e];
            float* row = agg + (size_t)d * kH + nlo;
#pragma unroll
            for (int nb = 0; nb < 8; ++nb) {
                float v = acc2[nb][r] + bias2[nb];
                (void)__hip_atomic_fetch_add(row + nb * 16, v,
                        __ATOMIC_RELAXED, __HIP_MEMORY_SCOPE_AGENT);
            }
        }
    }
}

// ---------------- node kernel: h = relu(concat(h,agg) @ Wu + bu) ------------
// Same cooperative weight staging; A from hbf (k<128) / fp32 agg (k>=128).
__global__ __launch_bounds__(256) void node_kernel(
    const unsigned short* __restrict__ hbfin,  // [N,128] bf16
    const float*          __restrict__ agg,    // [N,128] fp32
    const unsigned short* __restrict__ Wup,    // packed 8 k-slabs
    const float*          __restrict__ bu,
    float*                __restrict__ hout,
    unsigned short*       __restrict__ hbfout,
    int ntiles)
{
    __shared__ __align__(32) unsigned short wbuf[2][4096];        // 2 x 8KB

    const int tid  = threadIdx.x;
    const int wave = tid >> 5;
    const int lane = tid & 31;
    const int tile = blockIdx.x * BLK_WAVES + wave;
    const bool active = (tile < ntiles);       // N = 3125 full 16-row tiles
    const int nbase = tile * 16;
    const int kh  = lane >> 4;
    const int nlo = lane & 15;
    const int m   = lane & 15;
    const int node = active ? (nbase + m) : 0;

    v8f acc[8];
#pragma unroll
    for (int nb = 0; nb < 8; ++nb) acc[nb] = vzero8();

#pragma unroll
    for (int kb = 0; kb < 8; ++kb) {
        const int buf = kb & 1;
        stage_weights(Wup, &wbuf[buf][0], kb, tid);
        __syncthreads();
        if (active) {
            v16bf a;
            if (kb < 4) {            // h part, already bf16
                const unsigned short* ab =
                    hbfin + (size_t)node * kH + (kb & 3) * 32 + kh * 8;
                a = combine_a(*(const us8*)ab, *(const us8*)(ab + 16));
            } else {                 // agg part, fp32 -> bf16 in registers
                const float* fb = agg + (size_t)node * kH + (kb & 3) * 32 + kh * 8;
                float4 p0 = ((const float4*)fb)[0];
                float4 p1 = ((const float4*)fb)[1];
                float4 q0 = ((const float4*)(fb + 16))[0];
                float4 q1 = ((const float4*)(fb + 16))[1];
                us16 au;
                au[0]  = f2bf(p0.x); au[1]  = f2bf(p0.y);
                au[2]  = f2bf(p0.z); au[3]  = f2bf(p0.w);
                au[4]  = f2bf(p1.x); au[5]  = f2bf(p1.y);
                au[6]  = f2bf(p1.z); au[7]  = f2bf(p1.w);
                au[8]  = f2bf(q0.x); au[9]  = f2bf(q0.y);
                au[10] = f2bf(q0.z); au[11] = f2bf(q0.w);
                au[12] = f2bf(q1.x); au[13] = f2bf(q1.y);
                au[14] = f2bf(q1.z); au[15] = f2bf(q1.w);
                a = __builtin_bit_cast(v16bf, au);
            }
            v16bf bf[8];
#pragma unroll
            for (int nb = 0; nb < 8; ++nb) bf[nb] = lds_b_frag(&wbuf[buf][0], nb, lane);
#pragma unroll
            for (int nb = 0; nb < 8; ++nb) {
                acc[nb] = __builtin_amdgcn_wmma_f32_16x16x32_bf16(
                    false, a, false, bf[nb], (short)0, acc[nb], false, false);
            }
        }
    }
    if (!active) return;

#pragma unroll
    for (int nb = 0; nb < 8; ++nb) {
        float bias = bu[nb * 16 + nlo];
        int col = nb * 16 + nlo;
#pragma unroll
        for (int r = 0; r < 8; ++r) {
            int row = nbase + r + kh * 8;
            float x = fmaxf(acc[nb][r] + bias, 0.f);
            hout[(size_t)row * kH + col]   = x;
            hbfout[(size_t)row * kH + col] = f2bf(x);
        }
    }
}

// ---------------- logits head: [N,128] @ Wo[128,2] + bo ---------------------
__global__ __launch_bounds__(256) void logits_kernel(const float* __restrict__ h,
                                                     const float* __restrict__ Wo,
                                                     const float* __restrict__ bo,
                                                     float* __restrict__ out) {
    int n = blockIdx.x * 256 + threadIdx.x;
    if (n >= kN) return;
    float a0 = bo[0], a1 = bo[1];
    const float* hr = h + (size_t)n * kH;
#pragma unroll 4
    for (int k = 0; k < kH; ++k) {
        float x = hr[k];
        a0 += x * Wo[2 * k];
        a1 += x * Wo[2 * k + 1];
    }
    out[2 * n]     = a0;
    out[2 * n + 1] = a1;
}

// ---------------- host side -------------------------------------------------
extern "C" void kernel_launch(void* const* d_in, const int* in_sizes, int n_in,
                              void* d_out, int out_size, void* d_ws, size_t ws_size,
                              hipStream_t stream) {
    (void)in_sizes; (void)n_in; (void)out_size; (void)ws_size;

    const float*         x_nodes = (const float*)d_in[0];
    const int*           eidx    = (const int*)d_in[1];
    /* d_in[2] edge_attr unused */
    /* d_in[3] node_mask unused */
    const unsigned char* emask   = (const unsigned char*)d_in[4];
    const float*         W1      = (const float*)d_in[5];
    const float*         b1      = (const float*)d_in[6];
    const float*         W2      = (const float*)d_in[7];
    const float*         b2      = (const float*)d_in[8];
    const float*         Wu      = (const float*)d_in[9];
    const float*         bu      = (const float*)d_in[10];
    const float*         Wo      = (const float*)d_in[11];
    const float*         bo      = (const float*)d_in[12];
    float*               out     = (float*)d_out;

    // workspace carve-up (all offsets 256B aligned)
    char* ws = (char*)d_ws;
    float*          h    = (float*)(ws);                           // 25,600,000 B
    unsigned short* hbf  = (unsigned short*)(ws + 25600000);       // 12,800,000 B
    float*          agg  = (float*)(ws + 38400000);                // 25,600,000 B
    unsigned short* W1p  = (unsigned short*)(ws + 64000000);       //     65,536 B
    unsigned short* W2p  = (unsigned short*)(ws + 64065536);       //     32,768 B
    unsigned short* Wup  = (unsigned short*)(ws + 64098304);       //     65,536 B

    // 1) pack weights into WMMA B-fragment order (bf16, 8KB per k-step slab)
    pack_b_kernel<<<(8 * 8 * 32 + 255) / 256, 256, 0, stream>>>(W1, W1p, 8);
    pack_b_kernel<<<(4 * 8 * 32 + 255) / 256, 256, 0, stream>>>(W2, W2p, 4);
    pack_b_kernel<<<(8 * 8 * 32 + 255) / 256, 256, 0, stream>>>(Wu, Wup, 8);

    // 2) h = x_nodes (fp32 master + bf16 shadow)
    const size_t hcnt = (size_t)kN * kH;            // 6,400,000
    init_h_kernel<<<(int)((hcnt + 255) / 256), 256, 0, stream>>>(x_nodes, h, hbf);

    const int edge_blocks = kE / (16 * BLK_WAVES);                    // 12500
    const int ntiles      = kN / 16;                                  // 3125 exact
    const int node_blocks = (ntiles + BLK_WAVES - 1) / BLK_WAVES;     // 391

    // 3) message-passing iterations
    for (int it = 0; it < kITERS; ++it) {
        zero_kernel<<<(int)((hcnt + 255) / 256), 256, 0, stream>>>(agg, hcnt);
        edge_kernel<<<edge_blocks, 256, 0, stream>>>(hbf, eidx, emask,
                                                     W1p, b1, W2p, b2, agg);
        node_kernel<<<node_blocks, 256, 0, stream>>>(hbf, agg, Wup, bu,
                                                     h, hbf, ntiles);
    }

    // 4) output head
    logits_kernel<<<(kN + 255) / 256, 256, 0, stream>>>(h, Wo, bo, out);
}
